// GraphDecoder_19490561589318
// MI455X (gfx1250) — matrix operand order repaired and checked by reading, not compiled, not run on hardware
//
#include <hip/hip_runtime.h>
#include <hip/hip_bf16.h>
#include <math.h>

#define BB 4
#define NN 512
#define EE 128
#define HH 256
#define HF 128   // H/2

typedef __attribute__((ext_vector_type(16))) __bf16 v16bf;
typedef __attribute__((ext_vector_type(8)))  __bf16 v8bf;
typedef __attribute__((ext_vector_type(8)))  float  v8f;

// ---------------- precompute kernels (tiny cost) ----------------

// out[n][h] = bias[h] + sum_e emb[n][e] * W[e][h]       (f32)
__global__ void proj_kernel(const float* __restrict__ emb,
                            const float* __restrict__ W,
                            const float* __restrict__ bias,
                            float* __restrict__ out) {
  int n = blockIdx.x;
  int h = threadIdx.x;
  const float* er = emb + (size_t)n * EE;
  float s = bias[h];
#pragma unroll 8
  for (int e = 0; e < EE; ++e)
    s = fmaf(er[e], W[(size_t)e * HH + h], s);
  out[(size_t)n * HH + h] = s;
}

// out[n][h] = bf16( (bias?bias[h]:0) + sum_k in[n][k] * W[rowoff+k][h] )
__global__ void hid_kernel(const float* __restrict__ in,
                           const float* __restrict__ W,   // [2H, H]
                           int rowoff,
                           const float* __restrict__ bias, // may be null
                           __bf16* __restrict__ out) {
  int n = blockIdx.x;
  int h = threadIdx.x;
  const float* ir = in + (size_t)n * HH;
  float s = bias ? bias[h] : 0.f;
#pragma unroll 8
  for (int k = 0; k < HH; ++k)
    s = fmaf(ir[k], W[(size_t)(rowoff + k) * HH + h], s);
  out[(size_t)n * HH + h] = (__bf16)s;
}

// We2 [H, H/2] row-major -> W2T [H/2][H] bf16 (row = output channel)
__global__ void packW2T_kernel(const float* __restrict__ We2,
                               __bf16* __restrict__ W2T) {
  int t = blockIdx.x * blockDim.x + threadIdx.x;   // < HF*HH
  int nrow = t / HH;
  int k    = t % HH;
  W2T[t] = (__bf16)We2[(size_t)k * HF + nrow];
}

// Ww2 [H] f32 -> Ww2T [16][H] bf16: row0 = Ww2, rows 1..15 = 0.
__global__ void packWw2T_kernel(const float* __restrict__ Ww2,
                                __bf16* __restrict__ Ww2T) {
  int t = blockIdx.x * blockDim.x + threadIdx.x;   // < 16*HH
  int nrow = t / HH;
  int k    = t % HH;
  Ww2T[t] = (nrow == 0) ? (__bf16)Ww2[k] : (__bf16)0.0f;
}

// ---------------- main pairwise WMMA kernel ----------------
// block = 128 threads = 4 waves; wave handles (b, i, 16 consecutive j)
// grid = (NN/64, NN, BB)

// Packed bf16 ReLU: V_PK_MAX_NUM_BF16 on each dword (backend won't select
// it from generic fmaxnum, so emit directly).
__device__ __forceinline__ v8bf relu_pk(v8bf x, unsigned zero) {
  union { v8bf v; unsigned w[4]; } u;
  u.v = x;
#pragma unroll
  for (int d = 0; d < 4; ++d)
    asm("v_pk_max_num_bf16 %0, %1, %2" : "=v"(u.w[d]) : "v"(u.w[d]), "v"(zero));
  return u.v;
}

// Build one 16x32 bf16 A-fragment: relu(hs_row + ht_row) over a K-block.
// Adds lower to V_PK_ADD_BF16; ReLU via V_PK_MAX_NUM_BF16.
__device__ __forceinline__ v16bf build_afrag(const __bf16* hsrow,
                                             const __bf16* htrow,
                                             int ks, int hi, unsigned zero) {
  const int k0 = ks * 32 + hi * 8;   // lanes<16: K 0..7 ; lanes>=16: K 8..15
  const int k1 = k0 + 16;            // lanes<16: K 16..23; lanes>=16: K 24..31
  v8bf s0 = *(const v8bf*)(hsrow + k0);
  v8bf s1 = *(const v8bf*)(hsrow + k1);
  v8bf t0 = *(const v8bf*)(htrow + k0);
  v8bf t1 = *(const v8bf*)(htrow + k1);
  v8bf r0 = relu_pk(s0 + t0, zero);
  v8bf r1 = relu_pk(s1 + t1, zero);
  return __builtin_shufflevector(r0, r1,
      0,1,2,3,4,5,6,7,8,9,10,11,12,13,14,15);
}

// Load one 32x16 bf16 B-fragment from a [n][k] row-major matrix.
__device__ __forceinline__ v16bf load_bfrag(const __bf16* wr, int ks, int hi) {
  const int kb = ks * 32 + hi * 8;   // lanes<16: K 0..7,16..23 ; lanes>=16: K 8..15,24..31
  v8bf b0 = *(const v8bf*)(wr + kb);
  v8bf b1 = *(const v8bf*)(wr + kb + 16);
  return __builtin_shufflevector(b0, b1,
      0,1,2,3,4,5,6,7,8,9,10,11,12,13,14,15);
}

__global__ __launch_bounds__(128) void edge_kernel(
    const __bf16* __restrict__ hs_e, const __bf16* __restrict__ ht_e,
    const __bf16* __restrict__ hs_w, const __bf16* __restrict__ ht_w,
    const __bf16* __restrict__ W2T,  const __bf16* __restrict__ Ww2T,
    const float* __restrict__ be2, const float* __restrict__ We3,
    const float* __restrict__ be3, const float* __restrict__ bw2,
    float* __restrict__ out) {
  const int lane = threadIdx.x & 31;
  const int wave = threadIdx.x >> 5;
  const int b  = blockIdx.z;
  const int i  = blockIdx.y;
  const int j0 = (blockIdx.x * 4 + wave) * 16;
  const int m  = lane & 15;        // pair-in-tile (A row / B column)
  const int hi = lane >> 4;        // lane-half selector
  const size_t rowi = (size_t)(b * NN + i) * HH;
  const size_t rowj = (size_t)(b * NN + j0 + m) * HH;
  const size_t obase = ((size_t)b * NN + i) * NN + j0;
  unsigned zero = 0;
  asm("v_mov_b32 %0, 0" : "=v"(zero));   // packed-bf16 zero source

  // ================= weight MLP first (frees its registers early) ========
  // hw = relu(hs_w'[i] + ht_w[j]);  dot(hw, Ww2) via WMMA vs zero-padded B.
  {
    v16bf afw[8];
#pragma unroll
    for (int ks = 0; ks < 8; ++ks)
      afw[ks] = build_afrag(hs_w + rowi, ht_w + rowj, ks, hi, zero);

    v8f accw = {};
    const __bf16* wr = Ww2T + (size_t)m * HH;   // rows 1..15 are zero
#pragma unroll
    for (int ks = 0; ks < 8; ++ks) {
      v16bf bb = load_bfrag(wr, ks, hi);
      accw = __builtin_amdgcn_wmma_f32_16x16x32_bf16(
          false, afw[ks], false, bb, (short)0, accw, false, false);
    }
    // C layout: column N=0 lives in lane 0 (M=r) and lane 16 (M=8+r).
    const float b2 = bw2[0];
    float sp[8];
#pragma unroll
    for (int r = 0; r < 8; ++r) {
      float s = accw[r] + b2;
      sp[r] = (s > 20.f) ? s : log1pf(expf(s));
    }
    float* ow = out + (size_t)BB * NN * NN + obase;
    if (lane == 0) {
      *(float4*)(ow)     = make_float4(sp[0], sp[1], sp[2], sp[3]);
      *(float4*)(ow + 4) = make_float4(sp[4], sp[5], sp[6], sp[7]);
    } else if (lane == 16) {
      *(float4*)(ow + 8)  = make_float4(sp[0], sp[1], sp[2], sp[3]);
      *(float4*)(ow + 12) = make_float4(sp[4], sp[5], sp[6], sp[7]);
    }
  }

  // ================= edge MLP =================
  // he = relu(hs_e'[i] + ht_e[j]);  layer2: he @ We2 (K=256 -> 128 outs)
  v16bf afrag[8];
#pragma unroll
  for (int ks = 0; ks < 8; ++ks)
    afrag[ks] = build_afrag(hs_e + rowi, ht_e + rowj, ks, hi, zero);

  v8f acc[8] = {};
#pragma unroll
  for (int nt = 0; nt < 8; ++nt) {
    const __bf16* wr = W2T + (size_t)(nt * 16 + m) * HH;  // column n = nt*16+m
#pragma unroll
    for (int ks = 0; ks < 8; ++ks) {
      v16bf bb = load_bfrag(wr, ks, hi);
      acc[nt] = __builtin_amdgcn_wmma_f32_16x16x32_bf16(
          false, afrag[ks], false, bb, (short)0, acc[nt], false, false);
    }
  }

  // layer3: relu(acc + be2) . We3 + be3
  const int nl = lane & 15;            // C layout: lane -> N index
  float be2l[8], we3l[8];
#pragma unroll
  for (int nt = 0; nt < 8; ++nt) {
    be2l[nt] = be2[nt * 16 + nl];
    we3l[nt] = We3[nt * 16 + nl];
  }
  const float b3 = be3[0];
  float pr[8];
#pragma unroll
  for (int r = 0; r < 8; ++r) {        // C: VGPR r -> M = r (+8 for lanes>=16)
    float s = 0.f;
#pragma unroll
    for (int nt = 0; nt < 8; ++nt) {
      float v = acc[nt][r] + be2l[nt];
      s = fmaf(fmaxf(v, 0.f), we3l[nt], s);
    }
    pr[r] = s;
  }
#pragma unroll
  for (int r = 0; r < 8; ++r) {        // reduce over the 16 lanes holding row r
    float s = pr[r];
    s += __shfl_xor(s, 1, 32);
    s += __shfl_xor(s, 2, 32);
    s += __shfl_xor(s, 4, 32);
    s += __shfl_xor(s, 8, 32);
    pr[r] = s + b3;
  }
  if (lane == 0) {
    *(float4*)(out + obase)     = make_float4(pr[0], pr[1], pr[2], pr[3]);
    *(float4*)(out + obase + 4) = make_float4(pr[4], pr[5], pr[6], pr[7]);
  } else if (lane == 16) {
    *(float4*)(out + obase + 8)  = make_float4(pr[0], pr[1], pr[2], pr[3]);
    *(float4*)(out + obase + 12) = make_float4(pr[4], pr[5], pr[6], pr[7]);
  }
}

// ---------------- launcher ----------------

extern "C" void kernel_launch(void* const* d_in, const int* in_sizes, int n_in,
                              void* d_out, int out_size, void* d_ws, size_t ws_size,
                              hipStream_t stream) {
  const float* emb  = (const float*)d_in[0];
  const float* Wsrc = (const float*)d_in[1];
  const float* bsrc = (const float*)d_in[2];
  const float* Wtgt = (const float*)d_in[3];
  const float* btgt = (const float*)d_in[4];
  const float* We1  = (const float*)d_in[5];
  const float* be1  = (const float*)d_in[6];
  const float* We2  = (const float*)d_in[7];
  const float* be2  = (const float*)d_in[8];
  const float* We3  = (const float*)d_in[9];
  const float* be3  = (const float*)d_in[10];
  const float* Ww1  = (const float*)d_in[11];
  const float* bw1  = (const float*)d_in[12];
  const float* Ww2  = (const float*)d_in[13];
  const float* bw2  = (const float*)d_in[14];
  float* out = (float*)d_out;

  char* ws = (char*)d_ws;
  const size_t MB = 1024 * 1024;
  float*  src  = (float*)(ws);                  // 2 MB  [2048,256] f32
  float*  tgt  = (float*)(ws + 2 * MB);         // 2 MB
  __bf16* hs_e = (__bf16*)(ws + 4 * MB);        // 1 MB  [2048,256] bf16 (+be1)
  __bf16* ht_e = (__bf16*)(ws + 5 * MB);        // 1 MB
  __bf16* hs_w = (__bf16*)(ws + 6 * MB);        // 1 MB  (+bw1)
  __bf16* ht_w = (__bf16*)(ws + 7 * MB);        // 1 MB
  __bf16* W2T  = (__bf16*)(ws + 8 * MB);        // 64 KB [128,256] bf16
  __bf16* Ww2T = (__bf16*)(ws + 8 * MB + 65536);// 8 KB  [16,256] bf16

  proj_kernel<<<dim3(BB * NN), dim3(HH), 0, stream>>>(emb, Wsrc, bsrc, src);
  proj_kernel<<<dim3(BB * NN), dim3(HH), 0, stream>>>(emb, Wtgt, btgt, tgt);

  hid_kernel<<<dim3(BB * NN), dim3(HH), 0, stream>>>(src, We1, 0,  be1,     hs_e);
  hid_kernel<<<dim3(BB * NN), dim3(HH), 0, stream>>>(tgt, We1, HH, nullptr, ht_e);
  hid_kernel<<<dim3(BB * NN), dim3(HH), 0, stream>>>(src, Ww1, 0,  bw1,     hs_w);
  hid_kernel<<<dim3(BB * NN), dim3(HH), 0, stream>>>(tgt, Ww1, HH, nullptr, ht_w);

  packW2T_kernel<<<dim3((HF * HH) / 256), dim3(256), 0, stream>>>(We2, W2T);
  packWw2T_kernel<<<dim3((16 * HH) / 256), dim3(256), 0, stream>>>(Ww2, Ww2T);

  edge_kernel<<<dim3(NN / 64, NN, BB), dim3(128), 0, stream>>>(
      hs_e, ht_e, hs_w, ht_w, W2T, Ww2T, be2, We3, be3, bw2, out);
}
